// Model_83141976916212
// MI455X (gfx1250) — matrix operand order repaired and checked by reading, not compiled, not run on hardware
//
#include <hip/hip_runtime.h>

typedef __bf16 bf16;
typedef __attribute__((ext_vector_type(16))) __bf16 v16bf;
typedef __attribute__((ext_vector_type(8)))  float  v8f;

union Frag { v16bf v; unsigned int u[8]; };

#define S_LEN 512
#define BATCH 64
#define IDIM  128
#define HDIM  256
#define NLAY  6
#define GDIM  768           // 3*H
#define DDIM  512           // 2*H
#define MROWS (S_LEN*BATCH) // 32768
#define NGRP  4             // batch groups for the scan
#define GB    16            // batch rows per group

// 16-byte async DMA: global -> LDS (ASYNCcnt tracked, no VGPR data path)
__device__ __forceinline__ void async_copy16(void* lds_dst, const void* gsrc) {
    unsigned dst = (unsigned)(unsigned long long)lds_dst; // flat->LDS offset truncation
    asm volatile("global_load_async_to_lds_b128 %0, %1, off"
                 :: "v"(dst), "v"(gsrc) : "memory");
}
__device__ __forceinline__ void wait_async0() {
    asm volatile("s_wait_asynccnt 0x0" ::: "memory");
}

// ---------------------------------------------------------------- convert f32 -> bf16
__global__ void k_cvt(const float* __restrict__ s, bf16* __restrict__ d, long n) {
    long i  = (long)blockIdx.x * blockDim.x + threadIdx.x;
    long st = (long)gridDim.x * blockDim.x;
    for (; i < n; i += st) d[i] = (bf16)s[i];
}

// ---------------------------------------------------------------- gi = cur @ W^T + b
// A: MROWS x K bf16 row-major.  W: [dir][GDIM x K] bf16 row-major. gi: [dir][MROWS x GDIM] f32.
// 128x128 block tile, 8 waves, async double-buffered LDS staging.
__global__ void __launch_bounds__(256)
k_gi_gemm(const bf16* __restrict__ A, const bf16* __restrict__ W,
          const float* __restrict__ bias, float* __restrict__ gi, int K) {
    const int dir   = blockIdx.z;
    const int mBase = blockIdx.x * 128;
    const int nBase = blockIdx.y * 128;
    __shared__ bf16 As[2][128 * 32];
    __shared__ bf16 Bs[2][128 * 32];
    const int t     = threadIdx.x;
    const int lane  = t & 31;
    const int wv    = t >> 5;
    const int wm    = wv & 1;
    const int wn    = wv >> 1;
    const int l16   = lane & 15;
    const int khalf = lane >> 4;
    const bf16* Wd = W + (size_t)dir * GDIM * K;

    const v8f vzero = {0.f,0.f,0.f,0.f,0.f,0.f,0.f,0.f};
    v8f acc[4][2];
#pragma unroll
    for (int mt = 0; mt < 4; ++mt)
#pragma unroll
        for (int nt = 0; nt < 2; ++nt) acc[mt][nt] = vzero;

    auto stage = [&](int buf, int kb) {
#pragma unroll
        for (int c = 0; c < 2; ++c) {
            int idx = t * 16 + c * 8;
            int row = idx >> 5, col = idx & 31;
            async_copy16(&As[buf][row * 32 + col],
                         A + (size_t)(mBase + row) * K + kb + col);
            async_copy16(&Bs[buf][row * 32 + col],
                         Wd + (size_t)(nBase + row) * K + kb + col);
        }
    };

    stage(0, 0);
    int it = 0;
    for (int kb = 0; kb < K; kb += 32, ++it) {
        const int buf = it & 1;
        wait_async0();
        __syncthreads();                       // stage(buf) visible; prev compute done
        if (kb + 32 < K) stage(buf ^ 1, kb + 32);

        Frag a[4], b[2];
#pragma unroll
        for (int mt = 0; mt < 4; ++mt) {
            int r0 = wm * 64 + mt * 16 + l16;
#pragma unroll
            for (int v = 0; v < 8; ++v) {
                int k = (v < 4) ? (2 * v + 8 * khalf) : (16 + 2 * (v - 4) + 8 * khalf);
                a[mt].u[v] = *(const unsigned int*)&As[buf][r0 * 32 + k];
            }
        }
#pragma unroll
        for (int nt = 0; nt < 2; ++nt) {
            int n0 = wn * 32 + nt * 16 + l16;
#pragma unroll
            for (int v = 0; v < 8; ++v) {
                int k = (v < 4) ? (2 * v + 8 * khalf) : (16 + 2 * (v - 4) + 8 * khalf);
                b[nt].u[v] = *(const unsigned int*)&Bs[buf][n0 * 32 + k];
            }
        }
#pragma unroll
        for (int mt = 0; mt < 4; ++mt)
#pragma unroll
            for (int nt = 0; nt < 2; ++nt)
                acc[mt][nt] = __builtin_amdgcn_wmma_f32_16x16x32_bf16(
                    false, a[mt].v, false, b[nt].v, (short)0, acc[mt][nt], false, false);
    }

#pragma unroll
    for (int mt = 0; mt < 4; ++mt)
#pragma unroll
        for (int nt = 0; nt < 2; ++nt) {
            int   ncol = nBase + wn * 32 + nt * 16 + l16;
            float bv   = bias[dir * GDIM + ncol];
#pragma unroll
            for (int r = 0; r < 8; ++r) {
                int m = mBase + wm * 64 + mt * 16 + r + 8 * khalf;
                gi[(size_t)dir * MROWS * GDIM + (size_t)m * GDIM + ncol] = acc[mt][nt][r] + bv;
            }
        }
}

// ---------------------------------------------------------------- persistent GRU scan
// grid = (NGRP, 2dirs), block = 512 (16 waves). Each block scans 16 independent batch rows.
// whh B-fragments pinned in VGPRs (192/wave, loaded once). h: bf16 LDS (WMMA A) + f32 LDS
// (update path). gi[s] slice double-buffered in LDS via async DMA, prefetched 1 step ahead.
// Gate combine through a 48KB LDS gh buffer. 2 barriers per step.
__global__ void __launch_bounds__(512)
k_gru(const float* __restrict__ gi, const bf16* __restrict__ whh,
      const float* __restrict__ bhh, const float* __restrict__ h0,
      float* __restrict__ hn, bf16* __restrict__ nxt, int layer, int writeNext) {
    const int bg    = blockIdx.x;         // batch group
    const int dir   = blockIdx.y;
    const int t     = threadIdx.x;
    const int lane  = t & 31;
    const int wv    = t >> 5;             // 0..15
    const int l16   = lane & 15;
    const int khalf = lane >> 4;

    __shared__ bf16  hs[GB * HDIM];        //   8 KB  h (bf16, WMMA A source)
    __shared__ float hf[GB * HDIM];        //  16 KB  h (f32, update path)
    __shared__ float gh[GB * GDIM];        //  48 KB  per-step recurrent gates
    __shared__ float gib[2][GB * GDIM];    //  96 KB  gi slices (double buffered)

    const float* giD = gi  + (size_t)dir * MROWS * GDIM;
    const bf16*  whD = whh + (size_t)dir * GDIM * HDIM;
    const float* bhD = bhh + dir * GDIM;

    // ---- pin recurrent weights in registers: 3 N-tiles x 8 K-tiles per wave ----
    Frag bw[3][8];
#pragma unroll
    for (int i = 0; i < 3; ++i) {
        int ng = (wv * 3 + i) * 16 + l16;  // gate column 0..767
#pragma unroll
        for (int kt = 0; kt < 8; ++kt)
#pragma unroll
            for (int v = 0; v < 8; ++v) {
                int k = kt * 32 + ((v < 4) ? (2 * v + 8 * khalf) : (16 + 2 * (v - 4) + 8 * khalf));
                bw[i][kt].u[v] = *(const unsigned int*)&whD[(size_t)ng * HDIM + k];
            }
    }

    // pointwise mapping: thread t owns column jp (fixed) and rows bp+2i
    const int jp = t & 255;
    const int bp = t >> 8;
    const float biasG[3] = { bhD[jp], bhD[jp + 256], bhD[jp + 512] };

    // ---- init h ----
#pragma unroll
    for (int i = 0; i < 8; ++i) {
        int b = bp + 2 * i;
        float hv = h0[((size_t)(2 * layer + dir) * BATCH + bg * GB + b) * HDIM + jp];
        hf[b * HDIM + jp] = hv;
        hs[b * HDIM + jp] = (bf16)hv;
    }
    __syncthreads();

    auto stageGi = [&](int buf, int se) {
        const float* slice = giD + ((size_t)se * BATCH + bg * GB) * GDIM;
#pragma unroll
        for (int c = 0; c < 6; ++c) {
            int chunk = t + c * 512;   // 3072 x 16B chunks
            async_copy16(&gib[buf][chunk * 4], slice + (size_t)chunk * 4);
        }
    };

    stageGi(0, dir ? S_LEN - 1 : 0);

    const v8f vzero = {0.f,0.f,0.f,0.f,0.f,0.f,0.f,0.f};

    for (int s = 0; s < S_LEN; ++s) {
        const int se = dir ? (S_LEN - 1 - s) : s;
        const int p  = s & 1;

        // ---- recurrent GEMM: A from hs, B from pinned registers ----
        v8f acc[3];
#pragma unroll
        for (int i = 0; i < 3; ++i) acc[i] = vzero;

#pragma unroll
        for (int kt = 0; kt < 8; ++kt) {
            Frag a;
#pragma unroll
            for (int v = 0; v < 8; ++v) {
                int k = kt * 32 +
                        ((v < 4) ? (2 * v + 8 * khalf) : (16 + 2 * (v - 4) + 8 * khalf));
                a.u[v] = *(const unsigned int*)&hs[l16 * HDIM + k];
            }
#pragma unroll
            for (int i = 0; i < 3; ++i)
                acc[i] = __builtin_amdgcn_wmma_f32_16x16x32_bf16(
                    false, a.v, false, bw[i][kt].v, (short)0, acc[i], false, false);
        }
        // spill gates to LDS for cross-wave combine
#pragma unroll
        for (int i = 0; i < 3; ++i) {
            int ncol = (wv * 3 + i) * 16 + l16;
#pragma unroll
            for (int r = 0; r < 8; ++r) {
                int brow = r + 8 * khalf;
                gh[brow * GDIM + ncol] = acc[i][r];
            }
        }

        wait_async0();       // this wave's gi DMA (step s) done
        __syncthreads();     // gh + everyone's gi DMA visible; all hs reads done

        if (s + 1 < S_LEN) stageGi(p ^ 1, dir ? (S_LEN - 2 - s) : (s + 1));

        // ---- fused gates + state update ----
#pragma unroll
        for (int i = 0; i < 8; ++i) {
            int b  = bp + 2 * i;
            int gb = b * GDIM + jp;
            float gr = gib[p][gb], gz = gib[p][gb + 256], gn = gib[p][gb + 512];
            float ghr = gh[gb]       + biasG[0];
            float ghz = gh[gb + 256] + biasG[1];
            float ghn = gh[gb + 512] + biasG[2];
            float rg = 1.f / (1.f + __expf(-(gr + ghr)));
            float zg = 1.f / (1.f + __expf(-(gz + ghz)));
            float ng = tanhf(gn + rg * ghn);
            float hp = hf[b * HDIM + jp];
            float hv = (1.f - zg) * ng + zg * hp;
            hf[b * HDIM + jp] = hv;
            hs[b * HDIM + jp] = (bf16)hv;
            if (writeNext)
                nxt[((size_t)se * BATCH + bg * GB + b) * DDIM + dir * HDIM + jp] = (bf16)hv;
        }
        __syncthreads();     // h_new visible; gib[p]/gh reads done before reuse
    }

#pragma unroll
    for (int i = 0; i < 8; ++i) {
        int b = bp + 2 * i;
        hn[((size_t)(2 * layer + dir) * BATCH + bg * GB + b) * HDIM + jp] = hf[b * HDIM + jp];
    }
}

// ---------------------------------------------------------------- host driver
extern "C" void kernel_launch(void* const* d_in, const int* in_sizes, int n_in,
                              void* d_out, int out_size, void* d_ws, size_t ws_size,
                              hipStream_t stream) {
    const float* x     = (const float*)d_in[0];
    const float* h0    = (const float*)d_in[1];
    const float* w_ih0 = (const float*)d_in[2];
    const float* b_ih0 = (const float*)d_in[3];
    const float* w_ih  = (const float*)d_in[4];
    const float* b_ih  = (const float*)d_in[5];
    const float* w_hh  = (const float*)d_in[6];
    const float* b_hh  = (const float*)d_in[7];
    float* out = (float*)d_out;

    char*  ws     = (char*)d_ws;
    bf16*  curA   = (bf16*)(ws);                       //  32 MB activations (ping)
    bf16*  curB   = (bf16*)(ws + 33554432);            //  32 MB activations (pong)
    float* gi     = (float*)(ws + 67108864);           // 192 MB  gi[2][M*G] f32
    bf16*  wi_all = (bf16*)(ws + 268435456);           // 8.26 MB all input weights bf16
    bf16*  wh_all = (bf16*)(ws + 276692992);           // 4.72 MB all recurrent weights bf16

    // ---- up-front conversions (independent of layer outputs) ----
    k_cvt<<<2048, 256, 0, stream>>>(x, curA, (long)MROWS * IDIM);
    k_cvt<<<512,  256, 0, stream>>>(w_ih0, wi_all, (long)2 * GDIM * IDIM);
    k_cvt<<<2048, 256, 0, stream>>>(w_ih, wi_all + (size_t)2 * GDIM * IDIM,
                                    (long)(NLAY - 1) * 2 * GDIM * DDIM);
    k_cvt<<<2048, 256, 0, stream>>>(w_hh, wh_all, (long)NLAY * 2 * GDIM * HDIM);

    bf16* in = curA;
    bf16* nx = curB;
    for (int l = 0; l < NLAY; ++l) {
        const int    K    = (l == 0) ? IDIM : DDIM;
        const bf16*  wi_b = (l == 0) ? wi_all
                                     : (wi_all + (size_t)2 * GDIM * IDIM +
                                        (size_t)(l - 1) * 2 * GDIM * DDIM);
        const bf16*  wh_b = wh_all + (size_t)l * 2 * GDIM * HDIM;
        const float* bi_src = (l == 0) ? b_ih0 : (b_ih + (size_t)(l - 1) * 2 * GDIM);

        k_gi_gemm<<<dim3(MROWS / 128, GDIM / 128, 2), 256, 0, stream>>>(in, wi_b, bi_src, gi, K);

        k_gru<<<dim3(NGRP, 2), 512, 0, stream>>>(gi, wh_b, b_hh + (size_t)l * 2 * GDIM, h0,
                                                 out, nx, l, (l < NLAY - 1) ? 1 : 0);

        bf16* tmp = in; in = nx; nx = tmp;
    }
}
